// MambaVisionClassifier_87677462381070
// MI455X (gfx1250) — compile-verified, matrix operand order checked by reading, not run on hardware
//
#include <hip/hip_runtime.h>

typedef __attribute__((ext_vector_type(16))) __bf16 v16bf;
typedef __attribute__((ext_vector_type(8)))  float  v8f;
typedef __attribute__((ext_vector_type(4)))  int    v4i;

#define DI_ 2048
#define L_  512
#define B_  16
#define D_  1024

#if defined(__gfx1250__) && __has_builtin(__builtin_amdgcn_global_load_async_to_lds_b128)
#define USE_ASYNC_LDS 1
#else
#define USE_ASYNC_LDS 0
#endif

typedef __attribute__((address_space(1))) v4i gas_v4i;
typedef __attribute__((address_space(3))) v4i las_v4i;

__device__ __forceinline__ unsigned short f2bf(float f) {
    unsigned int u = __float_as_uint(f);
    unsigned int r = u + 0x7FFFu + ((u >> 16) & 1u);
    return (unsigned short)(r >> 16);
}

__device__ __forceinline__ void async_wait0() {
#if defined(__gfx1250__)
#if __has_builtin(__builtin_amdgcn_s_wait_asynccnt)
    __builtin_amdgcn_s_wait_asynccnt(0);
#else
    asm volatile("s_wait_asynccnt 0" ::: "memory");
#endif
#endif
}

union FragU { uint4 q[2]; v16bf v; };

// C[M,N] = A[M,K] @ B[N,K]^T (+ bias); A,B bf16 row-major, C fp32.
// BM=128, BN=64, BK=32, 256 threads = 8 waves (4Mx2N), 2x2 WMMA acc per wave.
// Tiles staged via GLOBAL_LOAD_ASYNC_TO_LDS_B128 (ASYNCcnt path) when available.
__global__ __launch_bounds__(256) void k_gemm_bf16(
    const unsigned short* __restrict__ A, int lda,
    const unsigned short* __restrict__ Bm, int ldb,
    float* __restrict__ C, int ldc,
    int M, int N, int K,
    const float* __restrict__ bias)
{
    __shared__ unsigned short As[128 * 40];   // padded stride 40 halves
    __shared__ unsigned short Bs[64 * 40];

    const int tid  = threadIdx.x;
    const int lane = tid & 31;
    const int w    = tid >> 5;
    const int wm   = w >> 1;        // 0..3
    const int wn   = w & 1;         // 0..1
    const int half = lane >> 4;     // 0/1
    const int mrow = lane & 15;

    const int bm = blockIdx.y * 128;
    const int bn = blockIdx.x * 64;

    // Per-thread staging chunks (16B = 8 halves each); OOB rows clamped so
    // EXEC stays uniform (garbage lands only in store-guarded outputs).
    // A tile: 128 rows x 4 chunks = 512 chunks -> 2/thread.
    int a_r[2], a_c[2], a_gm[2];
    #pragma unroll
    for (int i = 0; i < 2; ++i) {
        int ch = tid * 2 + i;
        a_r[i] = ch >> 2;
        a_c[i] = (ch & 3) * 8;
        int gm = bm + a_r[i];
        a_gm[i] = (gm < M) ? gm : (M - 1);
    }
    // B tile: 64 rows x 4 chunks = 256 chunks -> 1/thread.
    const int b_r = tid >> 2;
    const int b_c = (tid & 3) * 8;
    int gn = bn + b_r;
    const int b_gn = (gn < N) ? gn : (N - 1);

    v8f acc[2][2] = {};

    for (int k0 = 0; k0 < K; k0 += 32) {
#if USE_ASYNC_LDS
        #pragma unroll
        for (int i = 0; i < 2; ++i)
            __builtin_amdgcn_global_load_async_to_lds_b128(
                (gas_v4i*)(A + (size_t)a_gm[i] * lda + k0 + a_c[i]),
                (las_v4i*)(void*)&As[a_r[i] * 40 + a_c[i]], 0, 0);
        __builtin_amdgcn_global_load_async_to_lds_b128(
            (gas_v4i*)(Bm + (size_t)b_gn * ldb + k0 + b_c),
            (las_v4i*)(void*)&Bs[b_r * 40 + b_c], 0, 0);
        async_wait0();
#else
        #pragma unroll
        for (int i = 0; i < 2; ++i) {
            uint4 v = *(const uint4*)(A + (size_t)a_gm[i] * lda + k0 + a_c[i]);
            *(uint4*)&As[a_r[i] * 40 + a_c[i]] = v;
        }
        {
            uint4 v = *(const uint4*)(Bm + (size_t)b_gn * ldb + k0 + b_c);
            *(uint4*)&Bs[b_r * 40 + b_c] = v;
        }
#endif
        if (k0 + 32 < K)    // speculative prefetch of next A tile
            __builtin_prefetch(A + (size_t)a_gm[0] * lda + k0 + 32 + a_c[0], 0, 0);
        __syncthreads();

        // Fragments per ISA 16-bit layouts:
        // lane (t%16) = M/N index; element e: K = 16*(e/8) + 8*(t/16) + e%8
        FragU af[2], bfr[2];
        #pragma unroll
        for (int mi = 0; mi < 2; ++mi) {
            const unsigned short* p = &As[(wm * 32 + mi * 16 + mrow) * 40];
            af[mi].q[0] = *(const uint4*)(p + 8 * half);
            af[mi].q[1] = *(const uint4*)(p + 16 + 8 * half);
        }
        #pragma unroll
        for (int ni = 0; ni < 2; ++ni) {
            const unsigned short* p = &Bs[(wn * 32 + ni * 16 + mrow) * 40];
            bfr[ni].q[0] = *(const uint4*)(p + 8 * half);
            bfr[ni].q[1] = *(const uint4*)(p + 16 + 8 * half);
        }
        #pragma unroll
        for (int mi = 0; mi < 2; ++mi)
            #pragma unroll
            for (int ni = 0; ni < 2; ++ni)
                acc[mi][ni] = __builtin_amdgcn_wmma_f32_16x16x32_bf16(
                    false, af[mi].v, false, bfr[ni].v,
                    (short)0, acc[mi][ni], false, false);
        __syncthreads();
    }

    // Store: D layout lane n = t%16, vgpr r -> m = r + 8*(t/16)
    #pragma unroll
    for (int mi = 0; mi < 2; ++mi)
        #pragma unroll
        for (int ni = 0; ni < 2; ++ni) {
            int n  = bn + wn * 32 + ni * 16 + mrow;
            int mb = bm + wm * 32 + mi * 16 + half * 8;
            float bv = (bias != nullptr && n < N) ? bias[n] : 0.f;
            #pragma unroll
            for (int r = 0; r < 8; ++r) {
                int m = mb + r;
                if (m < M && n < N)
                    C[(size_t)m * ldc + n] = acc[mi][ni][r] + bv;
            }
        }
}

// fp32 -> bf16 (RNE), 4 elements/thread; n must be a multiple of 1024.
__global__ __launch_bounds__(256) void k_cvt_bf16(
    const float* __restrict__ in, unsigned short* __restrict__ out)
{
    size_t i = ((size_t)blockIdx.x * 256 + threadIdx.x) * 4;
    float4 v = *(const float4*)(in + i);
    ushort4 o;
    o.x = f2bf(v.x); o.y = f2bf(v.y); o.z = f2bf(v.z); o.w = f2bf(v.w);
    *(ushort4*)(out + i) = o;
}

// u = silu(causal_depthwise_conv4(xp) + conv_b); writes fp32 (scan) + bf16 (GEMM A)
__global__ __launch_bounds__(256) void k_conv_silu(
    const float* __restrict__ xp, const float* __restrict__ cw,
    const float* __restrict__ cb, float* __restrict__ u,
    unsigned short* __restrict__ ubf)
{
    size_t idx = (size_t)blockIdx.x * 256 + threadIdx.x;
    int c = (int)(idx % DI_);
    int l = (int)((idx / DI_) % L_);
    int b = (int)(idx / ((size_t)DI_ * L_));
    float acc = cb[c];
    #pragma unroll
    for (int k = 0; k < 4; ++k) {
        int ll = l - 3 + k;
        if (ll >= 0)
            acc += xp[((size_t)(b * L_ + ll)) * DI_ + c] * cw[c * 4 + k];
    }
    float s = acc / (1.f + __expf(-acc));
    u[idx] = s;
    ubf[idx] = f2bf(s);
}

__global__ __launch_bounds__(256) void k_softplus(float* __restrict__ t)
{
    size_t i = (size_t)blockIdx.x * 256 + threadIdx.x;
    float v = t[i];
    t[i] = (v > 20.f) ? v : log1pf(__expf(v));
}

// Selective scan over L; one thread per (b,d), 16 states in registers.
// Fuses D-skip + silu(z_last) gate; emits bf16 for the W_out GEMM.
__global__ __launch_bounds__(256) void k_scan(
    const float* __restrict__ dt, const float* __restrict__ u,
    const float* __restrict__ dbl, const float* __restrict__ A_log,
    const float* __restrict__ D_ssm, const float* __restrict__ z_last,
    unsigned short* __restrict__ y_gated_bf)
{
    const int tid = threadIdx.x;
    const int d   = blockIdx.x * 256 + tid;   // 0..2047
    const int b   = blockIdx.y;

    __shared__ float sB[16];
    __shared__ float sC[16];

    float A[16], h[16];
    #pragma unroll
    for (int s = 0; s < 16; ++s) {
        A[s] = -__expf(A_log[d * 16 + s]);
        h[s] = 0.f;
    }

    float uu = 0.f;
    for (int l = 0; l < L_; ++l) {
        size_t row = (size_t)(b * L_ + l);
        float dtv = dt[row * DI_ + d];
        uu        = u[row * DI_ + d];
        __syncthreads();
        if (tid < 16) sB[tid] = dbl[row * 96 + 64 + tid];
        if (l == L_ - 1 && tid < 16) sC[tid] = dbl[row * 96 + 80 + tid];
        __syncthreads();
        float du = dtv * uu;
        #pragma unroll
        for (int s = 0; s < 16; ++s)
            h[s] = h[s] * __expf(dtv * A[s]) + du * sB[s];
    }

    float y = 0.f;
    #pragma unroll
    for (int s = 0; s < 16; ++s) y += h[s] * sC[s];
    y += uu * D_ssm[d];
    float z = z_last[(size_t)b * DI_ + d];
    y *= z / (1.f + __expf(-z));
    y_gated_bf[(size_t)b * DI_ + d] = f2bf(y);
}

extern "C" void kernel_launch(void* const* d_in, const int* in_sizes, int n_in,
                              void* d_out, int out_size, void* d_ws, size_t ws_size,
                              hipStream_t stream) {
    const float* x      = (const float*)d_in[0];   // (16,512,1024)
    const float* W_in   = (const float*)d_in[1];   // (4096,1024)
    const float* conv_w = (const float*)d_in[2];   // (2048,4)
    const float* conv_b = (const float*)d_in[3];   // (2048)
    const float* W_x    = (const float*)d_in[4];   // (96,2048)
    const float* W_dt   = (const float*)d_in[5];   // (2048,64)
    const float* b_dt   = (const float*)d_in[6];   // (2048)
    const float* A_log  = (const float*)d_in[7];   // (2048,16)
    const float* D_ssm  = (const float*)d_in[8];   // (2048)
    const float* W_out  = (const float*)d_in[9];   // (1024,2048)
    const float* fc_w   = (const float*)d_in[10];  // (1000,1024)
    const float* fc_b   = (const float*)d_in[11];  // (1000)
    float* out = (float*)d_out;                    // (16,1000)

    const int M = B_ * L_;                          // 8192

    // ---- workspace carving (256B aligned) ----
    char* p = (char*)d_ws;
    auto carve = [&](size_t bytes) {
        char* q = p;
        p += (bytes + 255) & ~(size_t)255;
        return (void*)q;
    };
    unsigned short* xbf    = (unsigned short*)carve((size_t)M * D_ * 2);
    unsigned short* Winbf  = (unsigned short*)carve((size_t)4096 * D_ * 2);
    unsigned short* Wxbf   = (unsigned short*)carve((size_t)96 * DI_ * 2);
    unsigned short* Wdtbf  = (unsigned short*)carve((size_t)DI_ * 64 * 2);
    unsigned short* Woutbf = (unsigned short*)carve((size_t)D_ * DI_ * 2);
    unsigned short* fcwbf  = (unsigned short*)carve((size_t)1000 * D_ * 2);
    float*          xp     = (float*)carve((size_t)M * DI_ * 4);
    float*          u      = (float*)carve((size_t)M * DI_ * 4);
    unsigned short* ubf    = (unsigned short*)carve((size_t)M * DI_ * 2);
    float*          dbl    = (float*)carve((size_t)M * 96 * 4);
    unsigned short* dblbf  = (unsigned short*)carve((size_t)M * 96 * 2);
    float*          zlast  = (float*)carve((size_t)B_ * DI_ * 4);
    unsigned short* ygbf   = (unsigned short*)carve((size_t)B_ * DI_ * 2);
    float*          mlast  = (float*)carve((size_t)B_ * D_ * 4);
    unsigned short* mlbf   = (unsigned short*)carve((size_t)B_ * D_ * 2);
    float*          dt     = xp;                    // reuse: xp dead after conv

    dim3 blk(256);

    // 0) one-time fp32 -> bf16 conversions (n/1024 blocks each)
    k_cvt_bf16<<<(unsigned)((size_t)M * D_ / 1024), blk, 0, stream>>>(x, xbf);
    k_cvt_bf16<<<(unsigned)((size_t)4096 * D_ / 1024), blk, 0, stream>>>(W_in, Winbf);
    k_cvt_bf16<<<(unsigned)((size_t)96 * DI_ / 1024), blk, 0, stream>>>(W_x, Wxbf);
    k_cvt_bf16<<<(unsigned)((size_t)DI_ * 64 / 1024), blk, 0, stream>>>(W_dt, Wdtbf);
    k_cvt_bf16<<<(unsigned)((size_t)D_ * DI_ / 1024), blk, 0, stream>>>(W_out, Woutbf);
    k_cvt_bf16<<<(unsigned)((size_t)1000 * D_ / 1024), blk, 0, stream>>>(fc_w, fcwbf);

    // 1) xp = x @ W_in[:2048].T   (M=8192,N=2048,K=1024)
    k_gemm_bf16<<<dim3(DI_ / 64, M / 128), blk, 0, stream>>>(
        xbf, D_, Winbf, D_, xp, DI_, M, DI_, D_, nullptr);

    // 2) u = silu(conv(xp)+b) -> fp32 + bf16
    k_conv_silu<<<(unsigned)((size_t)M * DI_ / 256), blk, 0, stream>>>(
        xp, conv_w, conv_b, u, ubf);

    // 3) dbl = u @ W_x.T          (M=8192,N=96,K=2048)
    k_gemm_bf16<<<dim3(2, M / 128), blk, 0, stream>>>(
        ubf, DI_, Wxbf, DI_, dbl, 96, M, 96, DI_, nullptr);
    k_cvt_bf16<<<(unsigned)((size_t)M * 96 / 1024), blk, 0, stream>>>(dbl, dblbf);

    // 4) dt = dbl[:,:64] @ W_dt.T + b_dt   (M=8192,N=2048,K=64)
    k_gemm_bf16<<<dim3(DI_ / 64, M / 128), blk, 0, stream>>>(
        dblbf, 96, Wdtbf, 64, dt, DI_, M, DI_, 64, b_dt);

    // 5) dt = softplus(dt)
    k_softplus<<<(unsigned)((size_t)M * DI_ / 256), blk, 0, stream>>>(dt);

    // 6) z_last = x[:,L-1,:] @ W_in[2048:].T   (M=16,N=2048,K=1024)
    k_gemm_bf16<<<dim3(DI_ / 64, 1), blk, 0, stream>>>(
        xbf + (size_t)(L_ - 1) * D_, L_ * D_,
        Winbf + (size_t)DI_ * D_, D_, zlast, DI_, B_, DI_, D_, nullptr);

    // 7) selective scan + D-skip + silu(z) gate -> bf16
    k_scan<<<dim3(DI_ / 256, B_), blk, 0, stream>>>(
        dt, u, dbl, A_log, D_ssm, zlast, ygbf);

    // 8) mamba_last = ygate @ W_out.T   (M=16,N=1024,K=2048)
    k_gemm_bf16<<<dim3(D_ / 64, 1), blk, 0, stream>>>(
        ygbf, DI_, Woutbf, DI_, mlast, D_, B_, D_, DI_, nullptr);
    k_cvt_bf16<<<(unsigned)((size_t)B_ * D_ / 1024), blk, 0, stream>>>(mlast, mlbf);

    // 9) out = mamba_last @ fc_w.T + fc_b   (M=16,N=1000,K=1024)
    k_gemm_bf16<<<dim3((1000 + 63) / 64, 1), blk, 0, stream>>>(
        mlbf, D_, fcwbf, D_, out, 1000, B_, 1000, D_, fc_b);
}
